// R2LTransformer_86131274154479
// MI455X (gfx1250) — matrix operand order repaired and checked by reading, not compiled, not run on hardware
//
#include <hip/hip_runtime.h>
#include <cstdint>
#include <cstddef>

// ---------------------------------------------------------------------------
// R2L transformer for MI455X (gfx1250, wave32).
// All large GEMMs run on v_wmma_f32_16x16x32_bf16 (f32 accumulate).
// Wave tile 32x64 -> 8 WMMAs per K-step against 6 fragment loads.
// K is a template constant so all fragment loads use two base pointers plus
// compile-time immediate offsets (no address chains, no spills).
// Attention core (n=65, d=32) runs f32 VALU with K/V staged in LDS.
// ---------------------------------------------------------------------------

typedef __bf16 bf16_t;
typedef bf16_t v16bf __attribute__((ext_vector_type(16)));
typedef bf16_t v8bf  __attribute__((ext_vector_type(8)));
typedef float  v8f   __attribute__((ext_vector_type(8)));

namespace {

constexpr int   kDim    = 512;
constexpr int   kHeads  = 16;
constexpr int   kNwin   = 512;            // B * RH * RW
constexpr int   kNtok   = 65;             // 1 region + 64 local
constexpr int   kMrows  = kNwin * kNtok;  // 33280 (multiple of 64)
constexpr int   kNreg   = 512;            // B * RH * RW region rows
constexpr int   kDepth  = 4;
constexpr float kScale  = 0.17677669529663687f;  // 32^-0.5
constexpr float kEps    = 1e-3f;
constexpr size_t kLocalElems  = (size_t)8 * 64 * 64 * 512;  // 16777216
constexpr size_t kRegionElems = (size_t)8 * 8 * 8 * 512;    // 262144

enum { F_BIAS = 1, F_GELU = 2, F_RES = 4 };

// A matrix 16x32 fragment (CDNA5 ISA 7.12.2): lane = (h<<4)|r holds row r,
// this code packs K contiguously: [h*16,h*16+8) / [h*16+8,h*16+16) in VGPRs 0..3 / 4..7, B alike.
__device__ __forceinline__ v16bf make_a_frag(v8bf lo, v8bf hi) {
  v16bf f;
#pragma unroll
  for (int i = 0; i < 8; ++i) { f[i] = lo[i]; f[i + 8] = hi[i]; }
  return f;
}

// ---- GEMM: out = A[M,K] @ Bt[N,K]^T (+bias)(+gelu)(+res); f32 and/or bf16 --
// 8 waves / block in a 2x4 grid; each wave computes 32x64 (2x4 WMMA tiles);
// block tile 64x256. 8 WMMAs per K-step vs 6 fragment loads (21 FLOP/byte).
template <int K>
__global__ void gemm_wmma_kernel(const bf16_t* __restrict__ A,
                                 const bf16_t* __restrict__ Bt,
                                 float* out32, bf16_t* out16,
                                 const float* __restrict__ bias,
                                 int N, int flags) {
  int lane = threadIdx.x & 31;
  int wid  = threadIdx.x >> 5;
  int wm = blockIdx.y * 64  + (wid & 1) * 32;
  int wn = blockIdx.x * 256 + (wid >> 1) * 64;
  int r = lane & 15, h = lane >> 4;

  // Two per-lane base pointers; everything else is a compile-time offset.
  const bf16_t* pa = A  + (size_t)(wm + r) * K + h * 16;
  const bf16_t* pb = Bt + (size_t)(wn + r) * K + h * 16;

  v8f acc[2][4] = {};
#pragma unroll 4
  for (int k = 0; k < K; k += 32) {
    v16bf a0 = make_a_frag(*(const v8bf*)(pa + k),
                           *(const v8bf*)(pa + k + 8));
    v16bf a1 = make_a_frag(*(const v8bf*)(pa + 16 * K + k),
                           *(const v8bf*)(pa + 16 * K + k + 8));
    v16bf b0 = *(const v16bf*)(pb + k);
    v16bf b1 = *(const v16bf*)(pb + 16 * K + k);
    v16bf b2 = *(const v16bf*)(pb + 32 * K + k);
    v16bf b3 = *(const v16bf*)(pb + 48 * K + k);
    acc[0][0] = __builtin_amdgcn_wmma_f32_16x16x32_bf16(false, a0, false, b0, (short)0, acc[0][0], false, false);
    acc[0][1] = __builtin_amdgcn_wmma_f32_16x16x32_bf16(false, a0, false, b1, (short)0, acc[0][1], false, false);
    acc[0][2] = __builtin_amdgcn_wmma_f32_16x16x32_bf16(false, a0, false, b2, (short)0, acc[0][2], false, false);
    acc[0][3] = __builtin_amdgcn_wmma_f32_16x16x32_bf16(false, a0, false, b3, (short)0, acc[0][3], false, false);
    acc[1][0] = __builtin_amdgcn_wmma_f32_16x16x32_bf16(false, a1, false, b0, (short)0, acc[1][0], false, false);
    acc[1][1] = __builtin_amdgcn_wmma_f32_16x16x32_bf16(false, a1, false, b1, (short)0, acc[1][1], false, false);
    acc[1][2] = __builtin_amdgcn_wmma_f32_16x16x32_bf16(false, a1, false, b2, (short)0, acc[1][2], false, false);
    acc[1][3] = __builtin_amdgcn_wmma_f32_16x16x32_bf16(false, a1, false, b3, (short)0, acc[1][3], false, false);
  }

  int col = lane & 15, half = lane >> 4;
#pragma unroll
  for (int ti = 0; ti < 2; ++ti) {
#pragma unroll
    for (int tj = 0; tj < 4; ++tj) {
      int n = wn + tj * 16 + col;
      float bv = (flags & F_BIAS) ? bias[n] : 0.0f;
#pragma unroll
      for (int v = 0; v < 8; ++v) {
        int m = wm + ti * 16 + half * 8 + v;  // C/D layout: M = (lane>>4)*8 + v
        float x = acc[ti][tj][v] + bv;
        if (flags & F_GELU) x = 0.5f * x * (1.0f + erff(x * 0.70710678118654752f));
        size_t idx = (size_t)m * N + n;
        if (flags & F_RES) x += out32[idx];
        if (out32) out32[idx] = x;
        if (out16) out16[idx] = (bf16_t)x;
      }
    }
  }
}

// ---- LayerNorm (dim 512) + cast to bf16; one wave per row -------------------
__global__ void ln_cast_kernel(const float* __restrict__ x,
                               const float* __restrict__ g,
                               const float* __restrict__ b,
                               bf16_t* __restrict__ out, int M) {
  int wid = threadIdx.x >> 5, lane = threadIdx.x & 31;
  int row = blockIdx.x * 8 + wid;
  if (row >= M) return;
  const float* xr = x + (size_t)row * kDim;
  float xv[16], s = 0.f, ss = 0.f;
#pragma unroll
  for (int k = 0; k < 16; ++k) {
    float v = xr[lane + 32 * k];
    xv[k] = v; s += v; ss += v * v;
  }
#pragma unroll
  for (int off = 16; off; off >>= 1) {
    s  += __shfl_xor(s,  off, 32);
    ss += __shfl_xor(ss, off, 32);
  }
  float m   = s * (1.0f / kDim);
  float var = ss * (1.0f / kDim) - m * m;
  float r   = rsqrtf(var + kEps);
  bf16_t* orow = out + (size_t)row * kDim;
#pragma unroll
  for (int k = 0; k < 16; ++k) {
    int c = lane + 32 * k;
    orow[c] = (bf16_t)((xv[k] - m) * r * g[c] + b[c]);
  }
}

// ---- Attention core: one block per (group, head); one thread per query -----
template <int N>
__global__ void attn_kernel(const bf16_t* __restrict__ qkv, bf16_t* __restrict__ o,
                            const float* __restrict__ bias) {
  constexpr int kStride = 1536;  // qkv row stride
  __shared__ float Ks[N][32];
  __shared__ float Vs[N][32];
  int h = blockIdx.x & (kHeads - 1);
  int g = blockIdx.x / kHeads;
  int t = threadIdx.x;

  for (int idx = t; idx < N * 32; idx += blockDim.x) {
    int i = idx >> 5, c = idx & 31;
    const bf16_t* row = qkv + (size_t)(g * N + i) * kStride + h * 32;
    Ks[i][c] = (float)row[512 + c];
    Vs[i][c] = (float)row[1024 + c];
  }
  __syncthreads();

  if (t < N) {
    const bf16_t* qrow = qkv + (size_t)(g * N + t) * kStride + h * 32;
    float q[32];
#pragma unroll
    for (int c = 0; c < 32; ++c) q[c] = (float)qrow[c] * kScale;

    const float* brow = bias ? (bias + ((size_t)h * N + t) * N) : nullptr;
    float s[N], mx = -1e30f;
#pragma unroll
    for (int j = 0; j < N; ++j) {
      float d = 0.f;
#pragma unroll
      for (int c = 0; c < 32; ++c) d += q[c] * Ks[j][c];
      if (brow) d += brow[j];
      s[j] = d; mx = fmaxf(mx, d);
    }
    float sum = 0.f;
#pragma unroll
    for (int j = 0; j < N; ++j) { float e = expf(s[j] - mx); s[j] = e; sum += e; }
    float inv = 1.0f / sum;
    float ov[32];
#pragma unroll
    for (int c = 0; c < 32; ++c) ov[c] = 0.f;
#pragma unroll
    for (int j = 0; j < N; ++j) {
      float w = s[j] * inv;
#pragma unroll
      for (int c = 0; c < 32; ++c) ov[c] += w * Vs[j][c];
    }
    bf16_t* orow = o + (size_t)(g * N + t) * 512 + h * 32;
#pragma unroll
    for (int c = 0; c < 32; ++c) orow[c] = (bf16_t)ov[c];
  }
}

// ---- Swin relative-position bias [16][65][65] (row/col 0 zero-padded) ------
__global__ void bias_kernel(const float* __restrict__ emb, float* __restrict__ bias) {
  int idx = blockIdx.x * blockDim.x + threadIdx.x;
  if (idx >= kHeads * 65 * 65) return;
  int j = idx % 65, i = (idx / 65) % 65, h = idx / (65 * 65);
  float v = 0.f;
  if (i > 0 && j > 0) {
    int ii = i - 1, jj = j - 1;
    int xi = ii >> 3, yi = ii & 7, xj = jj >> 3, yj = jj & 7;
    int r = (xi - xj + 7) + (yi - yj + 7) * 15;
    v = emb[r * kHeads + h];
  }
  bias[idx] = v;
}

// ---- Weight transpose + cast: in[R][C] f32 -> out[C][R] bf16 ----------------
__global__ void transpose_cast_kernel(const float* __restrict__ in,
                                      bf16_t* __restrict__ out, int R, int C) {
  size_t idx = (size_t)blockIdx.x * blockDim.x + threadIdx.x;
  if (idx >= (size_t)R * C) return;
  int r = (int)(idx / C), c = (int)(idx % C);
  out[(size_t)c * R + r] = (bf16_t)in[idx];
}

__global__ void copy_f32_kernel(const float* __restrict__ in, float* __restrict__ out,
                                size_t n) {
  size_t idx = (size_t)blockIdx.x * blockDim.x + threadIdx.x;
  if (idx < n) out[idx] = in[idx];
}

// Layer 0: build x[w][p][c] from local_tokens (window partition) + region R
__global__ void gather_x_kernel(const float* __restrict__ local,
                                const float* __restrict__ R, float* __restrict__ X) {
  size_t idx = (size_t)blockIdx.x * blockDim.x + threadIdx.x;
  if (idx >= (size_t)kNwin * kNtok * kDim) return;
  int c = (int)(idx % kDim);
  int p = (int)((idx / kDim) % kNtok);
  int w = (int)(idx / ((size_t)kDim * kNtok));
  float v;
  if (p == 0) {
    v = R[(size_t)w * kDim + c];
  } else {
    int pp = p - 1, p1 = pp >> 3, p2 = pp & 7;
    int b = w >> 6, rh = (w >> 3) & 7, rw = w & 7;
    int y = rh * 8 + p1, xq = rw * 8 + p2;
    v = local[(((size_t)b * 64 + y) * 64 + xq) * kDim + c];
  }
  X[idx] = v;
}

__global__ void x0_from_R_kernel(float* __restrict__ X, const float* __restrict__ R) {
  int idx = blockIdx.x * blockDim.x + threadIdx.x;
  if (idx >= kNwin * kDim) return;
  int w = idx / kDim, c = idx % kDim;
  X[(size_t)w * kNtok * kDim + c] = R[idx];
}

__global__ void R_from_x0_kernel(const float* __restrict__ X, float* __restrict__ R) {
  int idx = blockIdx.x * blockDim.x + threadIdx.x;
  if (idx >= kNwin * kDim) return;
  int w = idx / kDim, c = idx % kDim;
  R[idx] = X[(size_t)w * kNtok * kDim + c];
}

// Final outputs: un-window-partition local, then region
__global__ void scatter_out_kernel(const float* __restrict__ X,
                                   const float* __restrict__ R,
                                   float* __restrict__ out) {
  size_t idx = (size_t)blockIdx.x * blockDim.x + threadIdx.x;
  if (idx >= kLocalElems + kRegionElems) return;
  if (idx < kLocalElems) {
    int c = (int)(idx % kDim);
    size_t t = idx / kDim;
    int xq = (int)(t % 64), y = (int)((t / 64) % 64), b = (int)(t / (64 * 64));
    int rh = y >> 3, p1 = y & 7, rw = xq >> 3, p2 = xq & 7;
    int w = b * 64 + rh * 8 + rw;
    int p = 1 + p1 * 8 + p2;
    out[idx] = X[((size_t)w * kNtok + p) * kDim + c];
  } else {
    out[idx] = R[idx - kLocalElems];
  }
}

}  // namespace

extern "C" void kernel_launch(void* const* d_in, const int* in_sizes, int n_in,
                              void* d_out, int out_size, void* d_ws, size_t ws_size,
                              hipStream_t stream) {
  (void)in_sizes; (void)n_in; (void)out_size; (void)ws_size;
  const float* local_tokens  = (const float*)d_in[0];
  const float* region_tokens = (const float*)d_in[1];
  const float* emb_table     = (const float*)d_in[2];
  const float* attn_ln_g     = (const float*)d_in[3];
  const float* attn_ln_b     = (const float*)d_in[4];
  const float* wqkv          = (const float*)d_in[5];
  const float* wout          = (const float*)d_in[6];
  const float* bout          = (const float*)d_in[7];
  const float* mlp_ln_g      = (const float*)d_in[8];
  const float* mlp_ln_b      = (const float*)d_in[9];
  const float* w1            = (const float*)d_in[10];
  const float* b1            = (const float*)d_in[11];
  const float* w2            = (const float*)d_in[12];
  const float* b2            = (const float*)d_in[13];
  float* out = (float*)d_out;

  char* ws = (char*)d_ws;
  size_t off = 0;
  auto take = [&](size_t bytes) -> char* {
    char* p = ws + off;
    off = (off + bytes + 255) & ~(size_t)255;
    return p;
  };

  bf16_t* wqkvT = (bf16_t*)take((size_t)kDepth * 1536 * 512 * 2);
  bf16_t* woutT = (bf16_t*)take((size_t)kDepth * 512 * 512 * 2);
  bf16_t* w1T   = (bf16_t*)take((size_t)kDepth * 2048 * 512 * 2);
  bf16_t* w2T   = (bf16_t*)take((size_t)kDepth * 512 * 2048 * 2);
  float*  biasB = (float*)take((size_t)kHeads * 65 * 65 * 4);
  float*  X     = (float*)take((size_t)kMrows * 512 * 4);
  bf16_t* H     = (bf16_t*)take((size_t)kMrows * 512 * 2);
  bf16_t* QKV   = (bf16_t*)take((size_t)kMrows * 1536 * 2);  // size % 256 == 0
  bf16_t* O     = (bf16_t*)take((size_t)kMrows * 512 * 2);   // contiguous after QKV
  bf16_t* G     = QKV;  // MLP hidden [kMrows,2048] aliases QKV∪O exactly
  float*  R     = (float*)take((size_t)kNreg * 512 * 4);
  bf16_t* rH    = (bf16_t*)take((size_t)kNreg * 512 * 2);
  bf16_t* rQKV  = (bf16_t*)take((size_t)kNreg * 1536 * 2);
  bf16_t* rO    = (bf16_t*)take((size_t)kNreg * 512 * 2);

  auto gemm = [&](const bf16_t* A, const bf16_t* Bt, float* o32, bf16_t* o16,
                  const float* bias, int M, int N, int K, int flags) {
    dim3 grid(N / 256, M / 64);
    if (K == 512)
      gemm_wmma_kernel<512><<<grid, 256, 0, stream>>>(A, Bt, o32, o16, bias, N, flags);
    else
      gemm_wmma_kernel<2048><<<grid, 256, 0, stream>>>(A, Bt, o32, o16, bias, N, flags);
  };

  // --- one-time (per launch) prep: bias table + bf16 transposed weights ---
  bias_kernel<<<(kHeads * 65 * 65 + 255) / 256, 256, 0, stream>>>(emb_table, biasB);
  for (int l = 0; l < kDepth; ++l) {
    transpose_cast_kernel<<<(512 * 1536 + 255) / 256, 256, 0, stream>>>(
        wqkv + (size_t)l * 512 * 1536, wqkvT + (size_t)l * 1536 * 512, 512, 1536);
    transpose_cast_kernel<<<(512 * 512 + 255) / 256, 256, 0, stream>>>(
        wout + (size_t)l * 512 * 512, woutT + (size_t)l * 512 * 512, 512, 512);
    transpose_cast_kernel<<<(512 * 2048 + 255) / 256, 256, 0, stream>>>(
        w1 + (size_t)l * 512 * 2048, w1T + (size_t)l * 2048 * 512, 512, 2048);
    transpose_cast_kernel<<<(2048 * 512 + 255) / 256, 256, 0, stream>>>(
        w2 + (size_t)l * 2048 * 512, w2T + (size_t)l * 512 * 2048, 2048, 512);
  }
  copy_f32_kernel<<<(int)((kRegionElems + 255) / 256), 256, 0, stream>>>(
      region_tokens, R, kRegionElems);

  const size_t xElems = (size_t)kNwin * kNtok * kDim;
  for (int l = 0; l < kDepth; ++l) {
    const bf16_t* wqkvT_l = wqkvT + (size_t)l * 1536 * 512;
    const bf16_t* woutT_l = woutT + (size_t)l * 512 * 512;
    const bf16_t* w1T_l   = w1T + (size_t)l * 2048 * 512;
    const bf16_t* w2T_l   = w2T + (size_t)l * 512 * 2048;
    const float* ag = attn_ln_g + l * 512, *ab = attn_ln_b + l * 512;
    const float* mg = mlp_ln_g + l * 512, *mb = mlp_ln_b + l * 512;
    const float* boutl = bout + l * 512;
    const float* b1l = b1 + l * 2048;
    const float* b2l = b2 + l * 512;

    // --- region self-attention (+residual) ---
    ln_cast_kernel<<<kNreg / 8, 256, 0, stream>>>(R, ag, ab, rH, kNreg);
    gemm(rH, wqkvT_l, nullptr, rQKV, nullptr, kNreg, 1536, 512, 0);
    attn_kernel<64><<<8 * kHeads, 96, 0, stream>>>(rQKV, rO, nullptr);
    gemm(rO, woutT_l, R, nullptr, boutl, kNreg, 512, 512, F_BIAS | F_RES);

    // --- assemble x: [region | window-partitioned local] ---
    if (l == 0)
      gather_x_kernel<<<(int)((xElems + 255) / 256), 256, 0, stream>>>(local_tokens, R, X);
    else
      x0_from_R_kernel<<<(kNwin * kDim + 255) / 256, 256, 0, stream>>>(X, R);

    // --- window attention with relative-position bias (+residual) ---
    ln_cast_kernel<<<kMrows / 8, 256, 0, stream>>>(X, ag, ab, H, kMrows);
    gemm(H, wqkvT_l, nullptr, QKV, nullptr, kMrows, 1536, 512, 0);
    attn_kernel<65><<<kNwin * kHeads, 96, 0, stream>>>(QKV, O, biasB);
    gemm(O, woutT_l, X, nullptr, boutl, kMrows, 512, 512, F_BIAS | F_RES);

    // --- MLP (+residual) ---
    ln_cast_kernel<<<kMrows / 8, 256, 0, stream>>>(X, mg, mb, H, kMrows);
    gemm(H, w1T_l, nullptr, G, b1l, kMrows, 2048, 512, F_BIAS | F_GELU);
    gemm(G, w2T_l, X, nullptr, b2l, kMrows, 512, 2048, F_BIAS | F_RES);

    // --- region = x[:,0] ---
    R_from_x0_kernel<<<(kNwin * kDim + 255) / 256, 256, 0, stream>>>(X, R);
  }

  scatter_out_kernel<<<(int)((kLocalElems + kRegionElems + 255) / 256), 256, 0, stream>>>(
      X, R, out);
}
